// FlashDeepseekLayer_64699387347303
// MI455X (gfx1250) — matrix operand order repaired and checked by reading, not compiled, not run on hardware
//
#include <hip/hip_runtime.h>
#include <hip/hip_bf16.h>

#define B_   2
#define S_   2048
#define D_   2048
#define H_   16
#define HD_  128
#define E_   16
#define TOPK_ 6
#define I_   1408
#define SI_  2816
#define T_   (B_ * S_)

typedef __bf16 bf16x16 __attribute__((ext_vector_type(16)));
typedef __bf16 bf16x8  __attribute__((ext_vector_type(8)));
typedef __bf16 bf16x4  __attribute__((ext_vector_type(4)));
typedef float  f32x8   __attribute__((ext_vector_type(8)));
typedef int    v4i     __attribute__((ext_vector_type(4)));

#define CONCAT16(lo, hi) \
    __builtin_shufflevector(lo, hi, 0, 1, 2, 3, 4, 5, 6, 7, 8, 9, 10, 11, 12, 13, 14, 15)

// ---- CDNA5 async global->LDS path (guarded; falls back to sync copies) ----
#if defined(__has_builtin)
#if __has_builtin(__builtin_amdgcn_global_load_async_to_lds_b128) && \
    __has_builtin(__builtin_amdgcn_s_wait_asynccnt)
#define USE_ASYNC 1
#endif
#endif
#ifndef USE_ASYNC
#define USE_ASYNC 0
#endif

#if USE_ASYNC
__device__ __forceinline__ void async_copy_b128(const void* g, void* l) {
    __attribute__((address_space(1))) v4i* gp =
        (__attribute__((address_space(1))) v4i*)(v4i*)const_cast<void*>(g);
    __attribute__((address_space(3))) v4i* lp =
        (__attribute__((address_space(3))) v4i*)(v4i*)l;
    __builtin_amdgcn_global_load_async_to_lds_b128(gp, lp, 0, 0);
}
__device__ __forceinline__ void wait_async0() {
    __builtin_amdgcn_s_wait_asynccnt(0);
}
#else
__device__ __forceinline__ void wait_async0() {}
#endif

__device__ __forceinline__ bf16x4 cvt4(const float4 f) {
    bf16x4 h;
    h[0] = (__bf16)f.x; h[1] = (__bf16)f.y;
    h[2] = (__bf16)f.z; h[3] = (__bf16)f.w;
    return h;
}

// ---------------------------------------------------------------------------
// bf16 WMMA GEMM with double-buffered (async) LDS staging.
//   A: bf16 [M,K]; B: bf16 [N,K] (all weights pre-transposed to N-major).
//   OB=true: store bf16; OB=false: fp32 with rowScale/residual/accumulate.
// LDS tile rows are 40 bf16 (80B) so every fragment is two ds_load_b128.
// M%128==0, N%128==0, K%32==0 guaranteed at launch sites.
// ---------------------------------------------------------------------------
template <bool OB>
__global__ __launch_bounds__(256) void gemm_bf16_wmma(
    const __bf16* __restrict__ A, const __bf16* __restrict__ Bm,
    void* __restrict__ Dv, int M, int N, int Kd,
    const float* __restrict__ residual,
    const float* __restrict__ rowScale, int rsStride, int accumulate) {
    __shared__ __align__(16) __bf16 As_raw[2][128 * 40];
    __shared__ __align__(16) __bf16 Bs_raw[2][128 * 40];

    const int tid  = threadIdx.x;
    const int lane = tid & 31;
    const int wid  = tid >> 5;
    const int waveM = (wid & 3) * 32;  // 4 waves along M, 32 rows each
    const int waveN = (wid >> 2) * 64; // 2 waves along N, 64 cols each
    const int row0 = blockIdx.y * 128;
    const int col0 = blockIdx.x * 128;
    const int mrow = lane & 15, khalf = lane >> 4;

    // stage one 128x32 bf16 tile pair; 512 16B segments each, 2 per thread
    auto stage = [&](int buf, int k0) {
        #pragma unroll
        for (int it = 0; it < 2; ++it) {
            int g = tid + it * 256;
            int r = g >> 2, seg = (g & 3) << 3;     // seg in elements (0,8,16,24)
            const __bf16* ga = &A[(size_t)(row0 + r) * Kd + k0 + seg];
            const __bf16* gb = &Bm[(size_t)(col0 + r) * Kd + k0 + seg];
            __bf16* la = &As_raw[buf][r * 40 + seg];
            __bf16* lb = &Bs_raw[buf][r * 40 + seg];
#if USE_ASYNC
            async_copy_b128(ga, la);
            async_copy_b128(gb, lb);
#else
            *(bf16x8*)la = *(const bf16x8*)ga;
            *(bf16x8*)lb = *(const bf16x8*)gb;
#endif
        }
    };

    const f32x8 fzero = {0.f, 0.f, 0.f, 0.f, 0.f, 0.f, 0.f, 0.f};
    f32x8 acc[2][4];
    #pragma unroll
    for (int mt = 0; mt < 2; ++mt)
        #pragma unroll
        for (int nt = 0; nt < 4; ++nt) acc[mt][nt] = fzero;

    stage(0, 0);
    const int nk = Kd >> 5;
    for (int kb = 0; kb < nk; ++kb) {
        wait_async0();
        __syncthreads();
        if (kb + 1 < nk) stage((kb + 1) & 1, (kb + 1) << 5);

        const bf16x8* As8 = (const bf16x8*)As_raw[kb & 1];
        const bf16x8* Bs8 = (const bf16x8*)Bs_raw[kb & 1];
        bf16x16 af[2], bfr[4];
        #pragma unroll
        for (int mt = 0; mt < 2; ++mt) {
            int r = waveM + mt * 16 + mrow;
            af[mt] = CONCAT16(As8[r * 5 + khalf], As8[r * 5 + 2 + khalf]);
        }
        #pragma unroll
        for (int nt = 0; nt < 4; ++nt) {
            int n = waveN + nt * 16 + mrow;
            bfr[nt] = CONCAT16(Bs8[n * 5 + khalf], Bs8[n * 5 + 2 + khalf]);
        }
        #pragma unroll
        for (int mt = 0; mt < 2; ++mt)
            #pragma unroll
            for (int nt = 0; nt < 4; ++nt)
                acc[mt][nt] = __builtin_amdgcn_wmma_f32_16x16x32_bf16(
                    false, af[mt], false, bfr[nt], (short)0, acc[mt][nt],
                    false, false);
    }

    // epilogue (C/D layout: VGPR r -> M = r + 8*(lane>=16); lane&15 -> N)
    const int nlane = lane & 15, half = lane >> 4;
    #pragma unroll
    for (int mt = 0; mt < 2; ++mt)
        #pragma unroll
        for (int nt = 0; nt < 4; ++nt)
            #pragma unroll
            for (int r = 0; r < 8; ++r) {
                int row = row0 + waveM + mt * 16 + half * 8 + r;
                int col = col0 + waveN + nt * 16 + nlane;
                size_t idx = (size_t)row * N + col;
                float v = acc[mt][nt][r];
                if (OB) {
                    ((__bf16*)Dv)[idx] = (__bf16)v;
                } else {
                    float* Df = (float*)Dv;
                    if (rowScale) v *= rowScale[(size_t)row * rsStride];
                    if (residual) v += residual[idx];
                    if (accumulate) Df[idx] += v;
                    else Df[idx] = v;
                }
            }
}

// ---------------------------------------------------------------------------
// Flash attention (causal), WMMA for QK^T and P*V. bf16 in, bf16 out.
// q/k/v: [T, H*HD] token-major. Block: 128 threads (4 waves), 64 q-rows.
//   Qs[64 q][136]   (hd contig)  -> A operand for QK^T
//   Ks[32 key][136] (hd contig)  -> B operand for QK^T (lane = key)
//   Vt[128 hd][40]  (key contig) -> B operand for P*V  (lane = hd)
//   Ps[64 q][40]    (key contig) -> A operand for P*V
// ---------------------------------------------------------------------------
__global__ __launch_bounds__(128) void flash_attn_k(
    const __bf16* __restrict__ Q, const __bf16* __restrict__ Kb,
    const __bf16* __restrict__ Vb, __bf16* __restrict__ O) {
    __shared__ __align__(16) __bf16 Qs_raw[64 * 136];
    __shared__ __align__(16) __bf16 Ks_raw[32 * 136];
    __shared__ __align__(16) __bf16 Vt_raw[128 * 40];
    __shared__ __align__(16) __bf16 Ps_raw[64 * 40];
    __shared__ float Sblk[64][36];
    __shared__ float rowm[64], rowl[64], ralpha[64];
    bf16x8* __restrict__ Qsw = (bf16x8*)Qs_raw;
    bf16x8* __restrict__ Ksw = (bf16x8*)Ks_raw;
    bf16x4* __restrict__ Vt4 = (bf16x4*)Vt_raw;
    const bf16x8* __restrict__ Qs8 = (const bf16x8*)Qs_raw;
    const bf16x8* __restrict__ Ks8 = (const bf16x8*)Ks_raw;
    const bf16x8* __restrict__ Vt8 = (const bf16x8*)Vt_raw;
    const bf16x8* __restrict__ Ps8 = (const bf16x8*)Ps_raw;

    const int tid = threadIdx.x, lane = tid & 31, wid = tid >> 5;
    const int bh = blockIdx.y;
    const int b = bh >> 4, h = bh & 15;
    const int q0 = blockIdx.x * 64;
    const int ldt = H_ * HD_;
    const size_t base = (size_t)b * S_ * ldt + (size_t)h * HD_;
    const int nlane = lane & 15, khalf = lane >> 4;

    // stage Q: 64x128 bf16, b128 copies (16 segs of 8 per row)
    #pragma unroll
    for (int it = 0; it < 8; ++it) {
        int g = tid + it * 128;           // 1024 segments
        int r = g >> 4, seg = g & 15;
        Qsw[r * 17 + seg] =
            *(const bf16x8*)&Q[base + (size_t)(q0 + r) * ldt + seg * 8];
    }
    if (tid < 64) { rowm[tid] = -1e30f; rowl[tid] = 0.f; }
    __syncthreads();

    // Q fragments: 4 K-steps over HD=128
    bf16x16 qf[4];
    {
        int m = wid * 16 + nlane;
        #pragma unroll
        for (int ks = 0; ks < 4; ++ks)
            qf[ks] = CONCAT16(Qs8[m * 17 + ks * 4 + khalf],
                              Qs8[m * 17 + ks * 4 + 2 + khalf]);
    }

    const f32x8 fzero = {0.f, 0.f, 0.f, 0.f, 0.f, 0.f, 0.f, 0.f};
    f32x8 oacc[8];
    #pragma unroll
    for (int nt = 0; nt < 8; ++nt) oacc[nt] = fzero;

    const int kend = q0 + 64;  // causal upper bound (block-uniform)
    for (int kc = 0; kc < kend; kc += 32) {
        __syncthreads();
        // stage K: 32x128 bf16 row-major
        #pragma unroll
        for (int it = 0; it < 4; ++it) {
            int g = tid + it * 128;       // 512 segments
            int r = g >> 4, seg = g & 15;
            Ksw[r * 17 + seg] =
                *(const bf16x8*)&Kb[base + (size_t)(kc + r) * ldt + seg * 8];
        }
        // stage V transposed: Vt[hd][key] via coalesced b16 row reads
        #pragma unroll
        for (int it = 0; it < 8; ++it) {
            int g = tid + it * 128;       // 1024 groups
            int hd = g & 127, kq = (g >> 7) << 2;
            bf16x4 hv;
            #pragma unroll
            for (int j = 0; j < 4; ++j)
                hv[j] = Vb[base + (size_t)(kc + kq + j) * ldt + hd];
            Vt4[hd * 10 + (kq >> 2)] = hv;
        }
        __syncthreads();

        // QK^T: M=16 (wave rows), N=32 keys, K=128
        f32x8 sacc[2] = {fzero, fzero};
        #pragma unroll
        for (int ks = 0; ks < 4; ++ks)
            #pragma unroll
            for (int nt = 0; nt < 2; ++nt) {
                int n = nt * 16 + nlane;  // key row
                bf16x16 bfr = CONCAT16(Ks8[n * 17 + ks * 4 + khalf],
                                       Ks8[n * 17 + ks * 4 + 2 + khalf]);
                sacc[nt] = __builtin_amdgcn_wmma_f32_16x16x32_bf16(
                    false, qf[ks], false, bfr, (short)0, sacc[nt], false,
                    false);
            }
        // scores -> LDS with scale + causal mask
        #pragma unroll
        for (int nt = 0; nt < 2; ++nt)
            #pragma unroll
            for (int r = 0; r < 8; ++r) {
                int rl = wid * 16 + khalf * 8 + r;
                int cl = nt * 16 + nlane;
                float sv = sacc[nt][r] * 0.08838834764831845f;
                if (kc + cl > q0 + rl) sv = -1e30f;
                Sblk[rl][cl] = sv;
            }
        __syncthreads();
        // online softmax per row; write P (bf16) key-contiguous
        if (tid < 64) {
            int rl = tid;
            float mold = rowm[rl];
            float mx = mold;
            for (int c = 0; c < 32; ++c) mx = fmaxf(mx, Sblk[rl][c]);
            float al = __expf(mold - mx);
            float sum = 0.f;
            for (int c = 0; c < 32; ++c) {
                float p = __expf(Sblk[rl][c] - mx);
                Ps_raw[rl * 40 + c] = (__bf16)p;
                sum += p;
            }
            rowm[rl] = mx;
            rowl[rl] = rowl[rl] * al + sum;
            ralpha[rl] = al;
        }
        __syncthreads();
        // rescale O, then O += P @ V
        {
            float al[8];
            #pragma unroll
            for (int r = 0; r < 8; ++r) al[r] = ralpha[wid * 16 + khalf * 8 + r];
            #pragma unroll
            for (int nt = 0; nt < 8; ++nt)
                #pragma unroll
                for (int r = 0; r < 8; ++r) oacc[nt][r] *= al[r];

            int m = wid * 16 + nlane;
            bf16x16 pf = CONCAT16(Ps8[m * 5 + khalf], Ps8[m * 5 + 2 + khalf]);
            #pragma unroll
            for (int nt = 0; nt < 8; ++nt) {
                int n = nt * 16 + nlane;  // hd
                bf16x16 vf = CONCAT16(Vt8[n * 5 + khalf],
                                      Vt8[n * 5 + 2 + khalf]);
                oacc[nt] = __builtin_amdgcn_wmma_f32_16x16x32_bf16(
                    false, pf, false, vf, (short)0, oacc[nt], false, false);
            }
        }
    }
    __syncthreads();
    #pragma unroll
    for (int nt = 0; nt < 8; ++nt)
        #pragma unroll
        for (int r = 0; r < 8; ++r) {
            int rl = wid * 16 + khalf * 8 + r;
            float inv = 1.0f / rowl[rl];
            O[base + (size_t)(q0 + rl) * ldt + nt * 16 + nlane] =
                (__bf16)(oacc[nt][r] * inv);
        }
}

// ---------------------------------------------------------------------------
// conversions: fp32 -> bf16 (straight, and 32x32 tiled transpose)
__global__ __launch_bounds__(256) void convert_k(const float* __restrict__ s,
                                                 __bf16* __restrict__ d,
                                                 size_t n4) {
    size_t i = (size_t)blockIdx.x * 256 + threadIdx.x;
    if (i < n4) ((bf16x4*)d)[i] = cvt4(((const float4*)s)[i]);
}

__global__ void convert_t_k(const float* __restrict__ s, __bf16* __restrict__ d,
                            int K, int N) {  // s: [K,N] fp32 -> d: [N,K] bf16
    __shared__ float tile[32][33];
    int kb = blockIdx.y * 32, nb = blockIdx.x * 32;
    for (int j = threadIdx.y; j < 32; j += 8)
        tile[j][threadIdx.x] = s[(size_t)(kb + j) * N + nb + threadIdx.x];
    __syncthreads();
    for (int j = threadIdx.y; j < 32; j += 8)
        d[(size_t)(nb + j) * K + kb + threadIdx.x] = (__bf16)tile[threadIdx.x][j];
}

// rmsnorm: fp32 in, bf16 out
__global__ __launch_bounds__(256) void rmsnorm_k(const float* __restrict__ x,
                                                 const float* __restrict__ w,
                                                 __bf16* __restrict__ o, int Dd) {
    __shared__ float red[256];
    int t = blockIdx.x;
    const float* xr = x + (size_t)t * Dd;
    float s = 0.f;
    for (int i = threadIdx.x; i < Dd; i += 256) {
        float v = xr[i];
        s += v * v;
    }
    red[threadIdx.x] = s;
    __syncthreads();
    for (int st = 128; st > 0; st >>= 1) {
        if (threadIdx.x < st) red[threadIdx.x] += red[threadIdx.x + st];
        __syncthreads();
    }
    float rs = rsqrtf(red[0] / (float)Dd + 1e-6f);
    for (int i = threadIdx.x; i < Dd; i += 256)
        o[(size_t)t * Dd + i] = (__bf16)(xr[i] * rs * w[i]);
}

// router: logits -> softmax -> top-6 renormalized combine weights cw[T,E]
__global__ __launch_bounds__(256) void gate_topk_k(const __bf16* __restrict__ hn,
                                                   const float* __restrict__ gw,
                                                   float* __restrict__ cw) {
    __shared__ float red[16][17];
    __shared__ float logits[16];
    int t = blockIdx.x;
    int e = threadIdx.x >> 4;
    int l = threadIdx.x & 15;
    const __bf16* xr = hn + (size_t)t * D_;
    const float* wr = gw + (size_t)e * D_;
    float s = 0.f;
    for (int i = l; i < D_; i += 16) s += (float)xr[i] * wr[i];
    red[e][l] = s;
    __syncthreads();
    if (l == 0) {
        float tot = 0.f;
        for (int i = 0; i < 16; ++i) tot += red[e][i];
        logits[e] = tot;
    }
    __syncthreads();
    if (threadIdx.x == 0) {
        float mx = -1e30f;
        for (int i = 0; i < E_; ++i) mx = fmaxf(mx, logits[i]);
        float probs[E_];
        float ssum = 0.f;
        for (int i = 0; i < E_; ++i) {
            probs[i] = __expf(logits[i] - mx);
            ssum += probs[i];
        }
        for (int i = 0; i < E_; ++i) probs[i] /= ssum;
        float out[E_];
        int used[E_];
        for (int i = 0; i < E_; ++i) { out[i] = 0.f; used[i] = 0; }
        float tsum = 0.f;
        for (int kk = 0; kk < TOPK_; ++kk) {
            int bi = 0;
            float bv = -1.f;
            for (int i = 0; i < E_; ++i)
                if (!used[i] && probs[i] > bv) { bv = probs[i]; bi = i; }
            used[bi] = 1;
            out[bi] = bv;
            tsum += bv;
        }
        float inv = 1.0f / tsum;
        for (int i = 0; i < E_; ++i) cw[(size_t)t * E_ + i] = out[i] * inv;
    }
}

__global__ void silu_mul_k(const __bf16* __restrict__ g,
                           const __bf16* __restrict__ u,
                           __bf16* __restrict__ o, size_t n) {
    size_t i = (size_t)blockIdx.x * blockDim.x + threadIdx.x;
    if (i < n) {
        float x = (float)g[i];
        float y = (float)u[i];
        o[i] = (__bf16)((x / (1.f + __expf(-x))) * y);
    }
}

// ---------------------------------------------------------------------------
extern "C" void kernel_launch(void* const* d_in, const int* in_sizes, int n_in,
                              void* d_out, int out_size, void* d_ws,
                              size_t ws_size, hipStream_t stream) {
    (void)in_sizes; (void)n_in; (void)out_size; (void)ws_size;
    const float* x    = (const float*)d_in[0];
    const float* ln1  = (const float*)d_in[1];
    const float* ln2  = (const float*)d_in[2];
    const float* wq   = (const float*)d_in[3];
    const float* wk   = (const float*)d_in[4];
    const float* wv   = (const float*)d_in[5];
    const float* wo   = (const float*)d_in[6];
    const float* gate = (const float*)d_in[7];
    const float* eg   = (const float*)d_in[8];
    const float* eu   = (const float*)d_in[9];
    const float* ed   = (const float*)d_in[10];
    const float* sg   = (const float*)d_in[11];
    const float* su   = (const float*)d_in[12];
    const float* sd   = (const float*)d_in[13];
    float* out = (float*)d_out;

    // workspace carve-up (256B aligned blocks)
    char* p = (char*)d_ws;
    auto alloc_f = [&](size_t n) {
        float* r = (float*)p;
        p += (n * sizeof(float) + 255) & ~(size_t)255;
        return r;
    };
    auto alloc_h = [&](size_t n) {
        __bf16* r = (__bf16*)p;
        p += (n * sizeof(__bf16) + 255) & ~(size_t)255;
        return r;
    };
    const size_t T = T_;
    float*  cw  = alloc_f(T * E_);
    __bf16* hn  = alloc_h(T * D_);
    __bf16* qb  = alloc_h(T * 2048);
    __bf16* kb  = alloc_h(T * 2048);
    __bf16* vb  = alloc_h(T * 2048);
    __bf16* ao  = alloc_h(T * 2048);
    __bf16* gb  = alloc_h(T * SI_);
    __bf16* ub  = alloc_h(T * SI_);
    __bf16* wqt = alloc_h((size_t)D_ * 2048);
    __bf16* wkt = alloc_h((size_t)D_ * 2048);
    __bf16* wvt = alloc_h((size_t)D_ * 2048);
    __bf16* wot = alloc_h((size_t)2048 * D_);
    __bf16* egc = alloc_h((size_t)E_ * I_ * D_);
    __bf16* euc = alloc_h((size_t)E_ * I_ * D_);
    __bf16* edc = alloc_h((size_t)E_ * D_ * I_);
    __bf16* sgc = alloc_h((size_t)SI_ * D_);
    __bf16* suc = alloc_h((size_t)SI_ * D_);
    __bf16* sdc = alloc_h((size_t)D_ * SI_);

    dim3 blk(256);
    dim3 tblk(32, 8);
    auto cvt = [&](const float* s, __bf16* d, size_t n) {
        size_t n4 = n >> 2;
        convert_k<<<(unsigned)((n4 + 255) / 256), 256, 0, stream>>>(s, d, n4);
    };
    // pre-convert weights to bf16 (attention weights transposed to [N,K])
    convert_t_k<<<dim3(2048 / 32, D_ / 32), tblk, 0, stream>>>(wq, wqt, D_, 2048);
    convert_t_k<<<dim3(2048 / 32, D_ / 32), tblk, 0, stream>>>(wk, wkt, D_, 2048);
    convert_t_k<<<dim3(2048 / 32, D_ / 32), tblk, 0, stream>>>(wv, wvt, D_, 2048);
    convert_t_k<<<dim3(D_ / 32, 2048 / 32), tblk, 0, stream>>>(wo, wot, 2048, D_);
    cvt(eg, egc, (size_t)E_ * I_ * D_);
    cvt(eu, euc, (size_t)E_ * I_ * D_);
    cvt(ed, edc, (size_t)E_ * D_ * I_);
    cvt(sg, sgc, (size_t)SI_ * D_);
    cvt(su, suc, (size_t)SI_ * D_);
    cvt(sd, sdc, (size_t)D_ * SI_);

    // attention block
    rmsnorm_k<<<(unsigned)T, 256, 0, stream>>>(x, ln1, hn, D_);
    dim3 gq(2048 / 128, T / 128);
    gemm_bf16_wmma<true><<<gq, blk, 0, stream>>>(hn, wqt, qb, T, 2048, D_,
                                                 nullptr, nullptr, 0, 0);
    gemm_bf16_wmma<true><<<gq, blk, 0, stream>>>(hn, wkt, kb, T, 2048, D_,
                                                 nullptr, nullptr, 0, 0);
    gemm_bf16_wmma<true><<<gq, blk, 0, stream>>>(hn, wvt, vb, T, 2048, D_,
                                                 nullptr, nullptr, 0, 0);
    flash_attn_k<<<dim3(S_ / 64, B_ * H_), 128, 0, stream>>>(qb, kb, vb, ao);
    // out = x + ao @ wo  (residual fused, fp32 out)
    gemm_bf16_wmma<false><<<dim3(D_ / 128, T / 128), blk, 0, stream>>>(
        ao, wot, out, T, D_, 2048, x, nullptr, 0, 0);

    // MoE block
    rmsnorm_k<<<(unsigned)T, 256, 0, stream>>>(out, ln2, hn, D_);
    gate_topk_k<<<(unsigned)T, 256, 0, stream>>>(hn, gate, cw);

    for (int e = 0; e < E_; ++e) {
        const __bf16* egw = egc + (size_t)e * I_ * D_;
        const __bf16* euw = euc + (size_t)e * I_ * D_;
        const __bf16* edw = edc + (size_t)e * D_ * I_;
        gemm_bf16_wmma<true><<<dim3(I_ / 128, T / 128), blk, 0, stream>>>(
            hn, egw, gb, T, I_, D_, nullptr, nullptr, 0, 0);
        gemm_bf16_wmma<true><<<dim3(I_ / 128, T / 128), blk, 0, stream>>>(
            hn, euw, ub, T, I_, D_, nullptr, nullptr, 0, 0);
        size_t n = (size_t)T * I_;
        silu_mul_k<<<(unsigned)((n + 255) / 256), 256, 0, stream>>>(gb, ub, gb, n);
        // out += cw[:,e] * (h @ ed^T)
        gemm_bf16_wmma<false><<<dim3(D_ / 128, T / 128), blk, 0, stream>>>(
            gb, edw, out, T, D_, I_, nullptr, cw + e, E_, 1);
    }

    // shared experts
    gemm_bf16_wmma<true><<<dim3(SI_ / 128, T / 128), blk, 0, stream>>>(
        hn, sgc, gb, T, SI_, D_, nullptr, nullptr, 0, 0);
    gemm_bf16_wmma<true><<<dim3(SI_ / 128, T / 128), blk, 0, stream>>>(
        hn, suc, ub, T, SI_, D_, nullptr, nullptr, 0, 0);
    size_t ns = (size_t)T * SI_;
    silu_mul_k<<<(unsigned)((ns + 255) / 256), 256, 0, stream>>>(gb, ub, gb, ns);
    gemm_bf16_wmma<false><<<dim3(D_ / 128, T / 128), blk, 0, stream>>>(
        gb, sdc, out, T, D_, SI_, nullptr, nullptr, 0, 1);
}
